// QMambaBlock_32762010534137
// MI455X (gfx1250) — matrix operand order repaired and checked by reading, not compiled, not run on hardware
//
#include <hip/hip_runtime.h>

// ---------------------------------------------------------------------------
// QMamba block for MI455X (gfx1250, wave32).  Big GEMMs use
// v_wmma_f32_16x16x32_bf16 (f32 accumulate); everything else is
// bandwidth-bound pointwise work or the latency-bound SSM scan.
// ---------------------------------------------------------------------------

typedef __attribute__((ext_vector_type(16))) __bf16 v16bf;
typedef __attribute__((ext_vector_type(8)))  float  v8f;

#define E_DIM   1024
#define DIN     2048
#define NSTATE  16
#define DTR     64
#define BATCH   2
#define LSEQ    1024
#define TOK     (BATCH * LSEQ)          // 2048 tokens

// ---------------------------------------------------------------------------
// TT-dense expansion:  W[a*m2+b, i*n2+j] = sum_r g1[a,i,r] * g2[r,b,j]
// stored TRANSPOSED in bf16:  wt[col*K + row]   (K = m1*m2 rows)
// so the GEMM's B operand is row-major over K -> dword-contiguous fragments.
// ---------------------------------------------------------------------------
__global__ void tt_expand_bt(const float* __restrict__ g1,
                             const float* __restrict__ g2,
                             __bf16* __restrict__ wt,
                             int m1, int n1, int R, int m2, int n2) {
  const int K = m1 * m2, N = n1 * n2;
  int idx = blockIdx.x * blockDim.x + threadIdx.x;
  if (idx >= K * N) return;
  int row = idx % K;            // consecutive threads -> consecutive rows
  int col = idx / K;
  int a = row / m2, b = row % m2;
  int i = col / n2, j = col % n2;
  float s = 0.f;
  #pragma unroll
  for (int r = 0; r < 16; ++r)
    s += g1[(a * n1 + i) * R + r] * g2[(r * m2 + b) * n2 + j];
  wt[(size_t)col * K + row] = (__bf16)s;
}

__global__ void f32_to_bf16(const float* __restrict__ src,
                            __bf16* __restrict__ dst, int n) {
  int i = blockIdx.x * blockDim.x + threadIdx.x;
  if (i < n) dst[i] = (__bf16)src[i];
}

__global__ void neg_exp(const float* __restrict__ a_log,
                        float* __restrict__ negA, int n) {
  int i = blockIdx.x * blockDim.x + threadIdx.x;
  if (i < n) negA[i] = -expf(a_log[i]);
}

// ---------------------------------------------------------------------------
// bf16 WMMA GEMM:  C[M,N](f32) = A[M,K](bf16,row-major) * BT[N,K](bf16)
// Compile-time tile: each wave owns (16*MSUB) x (16*NSUB) of C.
// MSUB*NSUB independent accumulators stay in fixed even-aligned VGPR blocks
// (no runtime predication around the WMMAs -> no accumulator mov chains).
// 8 waves per block tile M.  Requirements: M % (128*MSUB) == 0,
// N % (16*NSUB) == 0, K % 32 == 0 -- all launches below satisfy them.
// ---------------------------------------------------------------------------
template<int MSUB, int NSUB>
__global__ void __launch_bounds__(256)
gemm_bf16_wmma(const __bf16* __restrict__ A, const __bf16* __restrict__ BT,
               float* __restrict__ C, int N, int K) {
  const int lane = threadIdx.x & 31;
  const int wave = threadIdx.x >> 5;
  const int half = lane >> 4;          // 0: lanes 0-15, 1: lanes 16-31
  const int r    = lane & 15;

  const int mrow0 = (blockIdx.y * 8 + wave) * (16 * MSUB);
  const int ncol0 = blockIdx.x * (16 * NSUB);

  // A fragment rows (16-bit A 16x32 layout): lanes 0-15 K{0..7,16..23},
  // lanes 16-31 K{8..15,24..31}; fold the half-offset into the pointer.
  const __bf16* arow[MSUB];
  #pragma unroll
  for (int m = 0; m < MSUB; ++m)
    arow[m] = A + (size_t)(mrow0 + m * 16 + r) * K + half * 8;
  // B fragment rows (32x16): same K order as A (lanes 0-15 K{0..7,16..23}, lanes 16-31 K{8..15,24..31}).
  const __bf16* brow[NSUB];
  #pragma unroll
  for (int s = 0; s < NSUB; ++s)
    brow[s] = BT + (size_t)(ncol0 + s * 16 + r) * K + half * 8;

  v8f acc[MSUB][NSUB];
  const v8f vz = {0.f, 0.f, 0.f, 0.f, 0.f, 0.f, 0.f, 0.f};
  #pragma unroll
  for (int m = 0; m < MSUB; ++m)
    #pragma unroll
    for (int s = 0; s < NSUB; ++s) acc[m][s] = vz;

  union Frag { v16bf v; unsigned u[8]; };

  for (int kk = 0; kk < K; kk += 32) {
    Frag af[MSUB];
    #pragma unroll
    for (int m = 0; m < MSUB; ++m)
      #pragma unroll
      for (int i = 0; i < 4; ++i) {
        af[m].u[i]     = *(const unsigned*)(arow[m] + kk + 2 * i);       // b128
        af[m].u[i + 4] = *(const unsigned*)(arow[m] + kk + 16 + 2 * i);  // b128
      }
    Frag bfr[NSUB];
    #pragma unroll
    for (int s = 0; s < NSUB; ++s)
      #pragma unroll
      for (int i = 0; i < 4; ++i) {
        bfr[s].u[i]     = *(const unsigned*)(brow[s] + kk + 2 * i);       // b128
        bfr[s].u[i + 4] = *(const unsigned*)(brow[s] + kk + 16 + 2 * i);  // b128
      }
    #pragma unroll
    for (int m = 0; m < MSUB; ++m)
      #pragma unroll
      for (int s = 0; s < NSUB; ++s)
        acc[m][s] = __builtin_amdgcn_wmma_f32_16x16x32_bf16(
            false, af[m].v, false, bfr[s].v, (short)0, acc[m][s], false, false);
  }

  // C/D layout: VGPR v -> M = tile + v + 8*half, N = tile + r
  #pragma unroll
  for (int m = 0; m < MSUB; ++m)
    #pragma unroll
    for (int s = 0; s < NSUB; ++s) {
      const int col = ncol0 + s * 16 + r;
      #pragma unroll
      for (int v = 0; v < 8; ++v)
        C[(size_t)(mrow0 + m * 16 + v + 8 * half) * N + col] = acc[m][s][v];
    }
}

// ---------------------------------------------------------------------------
// Depthwise conv (K=4, SAME: pad 1 left / 2 right) + SiLU + z-gate.
// xz: (TOK, 4096) f32;  x_conv = cols [0,2048), z = cols [2048,4096)
// ---------------------------------------------------------------------------
__global__ void conv_silu_gate(const float* __restrict__ xz,
                               const float* __restrict__ filt,
                               const float* __restrict__ bias,
                               float* __restrict__ y_f,
                               __bf16* __restrict__ y_bf) {
  int idx = blockIdx.x * blockDim.x + threadIdx.x;
  if (idx >= TOK * DIN) return;
  int d = idx % DIN;
  int t = (idx / DIN) % LSEQ;
  int b = idx / (DIN * LSEQ);
  float s = bias[d];
  #pragma unroll
  for (int k = 0; k < 4; ++k) {
    int tt = t + k - 1;
    if (tt >= 0 && tt < LSEQ)
      s += xz[((size_t)(b * LSEQ + tt)) * 4096 + d] * filt[k * DIN + d];
  }
  float sig = 1.f / (1.f + expf(-s));
  float v = s * sig * xz[((size_t)(b * LSEQ + t)) * 4096 + 2048 + d];
  y_f[idx]  = v;
  y_bf[idx] = (__bf16)v;
}

// dtBC (TOK,96) -> dt_un bf16 (TOK,64)
__global__ void extract_dtun(const float* __restrict__ dtBC,
                             __bf16* __restrict__ dtun) {
  int i = blockIdx.x * blockDim.x + threadIdx.x;
  if (i >= TOK * DTR) return;
  dtun[i] = (__bf16)dtBC[(size_t)(i / DTR) * 96 + (i % DTR)];
}

__global__ void softplus_ip(float* __restrict__ x, int n) {
  int i = blockIdx.x * blockDim.x + threadIdx.x;
  if (i < n) {
    float v = x[i];
    x[i] = (v > 20.f) ? v : log1pf(expf(v));
  }
}

// ---------------------------------------------------------------------------
// SSM scan: one wave = two (b,d) channels; n in [0,16) mapped to lanes;
// y_t = sum_n h*C via 4-step shfl_xor reduction.  h-sum for mean(hs) term.
// ---------------------------------------------------------------------------
__global__ void __launch_bounds__(256)
ssm_scan(const float* __restrict__ dt, const float* __restrict__ y,
         const float* __restrict__ dtBC, const float* __restrict__ negA,
         const float* __restrict__ Dp, float* __restrict__ ys,
         float* __restrict__ hsum) {
  const int lane = threadIdx.x & 31;
  const int n    = lane & 15;
  const int pair = lane >> 4;
  const int wgl  = (blockIdx.x * blockDim.x + threadIdx.x) >> 5;
  const int ch   = wgl * 2 + pair;           // 0 .. BATCH*DIN-1 (exact)
  const int b    = ch >> 11;                 // / DIN
  const int d    = ch & (DIN - 1);

  const float a  = negA[d * NSTATE + n];
  const float Dd = Dp[d];
  float h = 0.f, hs = 0.f;

  for (int t = 0; t < LSEQ; ++t) {
    const size_t row = (size_t)(b * LSEQ + t);
    const float dtv = dt[row * DIN + d];
    const float u   = y[row * DIN + d];
    const float Bt  = dtBC[row * 96 + 64 + n];
    const float Ct  = dtBC[row * 96 + 80 + n];
    h = expf(dtv * a) * h + dtv * Bt * u;
    hs += h;
    float p = h * Ct;
    p += __shfl_xor(p, 1);
    p += __shfl_xor(p, 2);
    p += __shfl_xor(p, 4);
    p += __shfl_xor(p, 8);
    if (n == 0) ys[row * DIN + d] = p + Dd * u;
  }
  hs += __shfl_xor(hs, 1);
  hs += __shfl_xor(hs, 2);
  hs += __shfl_xor(hs, 4);
  hs += __shfl_xor(hs, 8);
  if (n == 0) atomicAdd(hsum, hs);
}

__global__ void add_mean_bf16(const float* __restrict__ ys,
                              const float* __restrict__ hsum,
                              __bf16* __restrict__ ssm_bf, int nelem) {
  int i = blockIdx.x * blockDim.x + threadIdx.x;
  if (i >= nelem) return;
  const float bias = hsum[0] * (1e-8f / 67108864.0f);  // / (L*B*DIN*N)
  ssm_bf[i] = (__bf16)(ys[i] + bias);
}

// ---------------------------------------------------------------------------
extern "C" void kernel_launch(void* const* d_in, const int* in_sizes, int n_in,
                              void* d_out, int out_size, void* d_ws, size_t ws_size,
                              hipStream_t stream) {
  (void)in_sizes; (void)n_in; (void)out_size; (void)ws_size;
  const float* x      = (const float*)d_in[0];
  const float* in_g1  = (const float*)d_in[1];
  const float* in_g2  = (const float*)d_in[2];
  const float* x_g1   = (const float*)d_in[3];
  const float* x_g2   = (const float*)d_in[4];
  const float* dt_g1  = (const float*)d_in[5];
  const float* dt_g2  = (const float*)d_in[6];
  const float* out_g1 = (const float*)d_in[7];
  const float* out_g2 = (const float*)d_in[8];
  const float* convf  = (const float*)d_in[9];
  const float* convb  = (const float*)d_in[10];
  const float* A_log  = (const float*)d_in[11];
  const float* Dp     = (const float*)d_in[12];
  float* out = (float*)d_out;

  // ---- workspace carve-up (256B aligned) ----
  size_t off = 0;
  auto carve = [&](size_t bytes) -> char* {
    char* p = (char*)d_ws + off;
    off += (bytes + 255) & ~(size_t)255;
    return p;
  };
  __bf16* wt_in  = (__bf16*)carve((size_t)1024 * 4096 * 2);
  __bf16* wt_x   = (__bf16*)carve((size_t)2048 * 96   * 2);
  __bf16* wt_dt  = (__bf16*)carve((size_t)64   * 2048 * 2);
  __bf16* wt_out = (__bf16*)carve((size_t)2048 * 1024 * 2);
  __bf16* x_bf   = (__bf16*)carve((size_t)TOK * E_DIM * 2);
  float*  xz     = (float*) carve((size_t)TOK * 4096  * 4);   // reused later
  float*  y_f    = (float*) carve((size_t)TOK * DIN   * 4);
  __bf16* y_bf   = (__bf16*)carve((size_t)TOK * DIN   * 2);
  float*  dtBC   = (float*) carve((size_t)TOK * 96    * 4);
  __bf16* dtun   = (__bf16*)carve((size_t)TOK * DTR   * 2);
  float*  negA   = (float*) carve((size_t)DIN * NSTATE * 4);
  __bf16* ssm_bf = (__bf16*)carve((size_t)TOK * DIN   * 2);
  float*  hsum   = (float*) carve(256);
  // alias the dead xz region once conv has consumed it:
  float* dt_f = xz;                                   // (TOK, DIN) f32, 16MB
  float* ys   = xz + (size_t)TOK * DIN;               // (TOK, DIN) f32, 16MB

  const int TB = 256;
  auto blks = [](size_t n, int tb) { return (unsigned)((n + tb - 1) / tb); };

  // 1) expand TT weights (bf16, transposed)
  tt_expand_bt<<<blks((size_t)1024 * 4096, TB), TB, 0, stream>>>(in_g1, in_g2, wt_in, 32, 64, 16, 32, 64);
  tt_expand_bt<<<blks((size_t)2048 * 96,   TB), TB, 0, stream>>>(x_g1, x_g2, wt_x, 32, 8, 16, 64, 12);
  tt_expand_bt<<<blks((size_t)64 * 2048,   TB), TB, 0, stream>>>(dt_g1, dt_g2, wt_dt, 8, 32, 16, 8, 64);
  tt_expand_bt<<<blks((size_t)2048 * 1024, TB), TB, 0, stream>>>(out_g1, out_g2, wt_out, 32, 32, 16, 64, 32);
  neg_exp<<<blks(DIN * NSTATE, TB), TB, 0, stream>>>(A_log, negA, DIN * NSTATE);

  // 2) x -> bf16 ; in-proj GEMM: (2048x1024)x(1024x4096) -> xz
  f32_to_bf16<<<blks((size_t)TOK * E_DIM, TB), TB, 0, stream>>>(x, x_bf, TOK * E_DIM);
  gemm_bf16_wmma<2, 4><<<dim3(4096 / 64, TOK / 256), TB, 0, stream>>>(x_bf, wt_in, xz, 4096, E_DIM);

  // 3) depthwise conv + silu + gate
  conv_silu_gate<<<blks((size_t)TOK * DIN, TB), TB, 0, stream>>>(xz, convf, convb, y_f, y_bf);

  // 4) x-proj GEMM: (2048x2048)x(2048x96) -> dtBC   (96 = 3 * 32)
  gemm_bf16_wmma<2, 2><<<dim3(96 / 32, TOK / 256), TB, 0, stream>>>(y_bf, wt_x, dtBC, 96, DIN);
  extract_dtun<<<blks((size_t)TOK * DTR, TB), TB, 0, stream>>>(dtBC, dtun);

  // 5) dt-proj GEMM: (2048x64)x(64x2048) -> dt_f (aliases xz) ; softplus
  gemm_bf16_wmma<2, 4><<<dim3(2048 / 64, TOK / 256), TB, 0, stream>>>(dtun, wt_dt, dt_f, DIN, DTR);
  softplus_ip<<<blks((size_t)TOK * DIN, TB), TB, 0, stream>>>(dt_f, TOK * DIN);

  // 6) selective scan
  hipMemsetAsync(hsum, 0, sizeof(float), stream);
  ssm_scan<<<(BATCH * DIN * 16) / TB, TB, 0, stream>>>(dt_f, y_f, dtBC, negA, Dp, ys, hsum);

  // 7) + mean(hs)*1e-8, bf16 ; out-proj GEMM: (2048x2048)x(2048x1024) -> out
  add_mean_bf16<<<blks((size_t)TOK * DIN, TB), TB, 0, stream>>>(ys, hsum, ssm_bf, TOK * DIN);
  gemm_bf16_wmma<2, 4><<<dim3(1024 / 64, TOK / 256), TB, 0, stream>>>(ssm_bf, wt_out, out, E_DIM, DIN);
}
